// GNNLayer_58634893525192
// MI455X (gfx1250) — compile-verified, matrix operand order checked by reading, not run on hardware
//
#include <hip/hip_runtime.h>
#include <hip/hip_bf16.h>

// ---------------------------------------------------------------------------
// GNN message-passing layer for MI455X (gfx1250), wave32 + WMMA bf16.
//   edge MLP:  silu(silu([x[src]||ea] @ W1^T + b1) @ W2^T + b2)  -> scatter-add
//   GRU cell:  gates via agg@W_ih^T and x@W_hh^T (both WMMA), fused epilogue
// ---------------------------------------------------------------------------

typedef __bf16 bf16_t;
typedef __bf16 v16bf __attribute__((ext_vector_type(16)));
typedef __bf16 v8bf  __attribute__((ext_vector_type(8)));
typedef __bf16 v4bf  __attribute__((ext_vector_type(4)));
typedef float  v8f   __attribute__((ext_vector_type(8)));

#define ND    128      // node dim
#define ED    32       // edge dim
#define KIN   160      // node+edge concat
#define HID   128      // hidden
#define AS_IN 176      // LDS row stride (halfs) for 160-wide A tile (16B-aligned, pad vs banks)
#define AS_H  144      // LDS row stride (halfs) for 128-wide A tile

__device__ __forceinline__ v8f wmma_bf16(v16bf a, v16bf b, v8f c) {
  // D(f32 16x16) = A(bf16 16x32) * B(bf16 32x16) + C
  return __builtin_amdgcn_wmma_f32_16x16x32_bf16(false, a, false, b, (short)0, c,
                                                 false, false);
}

// Assemble the per-lane A fragment for one K-chunk of 32 from a row-major
// bf16 LDS tile. 16-bit A layout: lanes 0-15 = rows (K-half 0), lanes 16-31 =
// rows (K-half 1); each lane holds two contiguous runs of 8 K-values.
__device__ __forceinline__ v16bf load_afrag(const bf16_t* base, int strideHalf,
                                            int lane, int kc) {
  const int row = lane & 15;
  const int g   = lane >> 4;
  const bf16_t* p = base + row * strideHalf + kc * 32 + g * 8;
  v8bf lo = *(const v8bf*)p;          // K = kc*32 + g*8 + [0..7]
  v8bf hi = *(const v8bf*)(p + 16);   // K = kc*32 + 16 + g*8 + [0..7]
  v16bf a;
#pragma unroll
  for (int i = 0; i < 8; ++i) { a[i] = lo[i]; a[8 + i] = hi[i]; }
  return a;
}

// Fast nonlinearities: v_exp_f32 + v_rcp_f32, no IEEE divide sequence.
__device__ __forceinline__ float fast_sigmoid(float x) {
  return __builtin_amdgcn_rcpf(1.f + __expf(-x));
}
__device__ __forceinline__ float fast_silu(float x) { return x * fast_sigmoid(x); }
__device__ __forceinline__ float fast_tanh(float x) {
  return 2.f * fast_sigmoid(2.f * x) - 1.f;
}

__device__ __forceinline__ v4bf cvt4(float4 v) {
  v4bf c;
  c[0] = (bf16_t)v.x; c[1] = (bf16_t)v.y; c[2] = (bf16_t)v.z; c[3] = (bf16_t)v.w;
  return c;
}

// ---------------------------------------------------------------------------
// Pack a row-major f32 weight W[Nout x K] (used as B = W^T, i.e. B[k][n] =
// W[n][k]) into bf16 per-lane WMMA B-fragments: out[((nt*kchunks+kc)*32+l)*16+h]
// so the hot kernels fetch one contiguous 32B vector per lane per WMMA.
// ---------------------------------------------------------------------------
__global__ void pack_weight_frags(const float* __restrict__ W, int K, int kchunks,
                                  int total, bf16_t* __restrict__ out) {
  for (int idx = blockIdx.x * blockDim.x + threadIdx.x; idx < total;
       idx += gridDim.x * blockDim.x) {
    const int h  = idx & 15;
    const int l  = (idx >> 4) & 31;
    const int kc = (idx >> 9) % kchunks;
    const int nt = (idx >> 9) / kchunks;
    const int n  = nt * 16 + (l & 15);
    const int g  = l >> 4;
    const int kk = (h < 8) ? (g * 8 + h) : (16 + g * 8 + (h - 8));
    const int k  = kc * 32 + kk;
    out[idx] = (k < K) ? (bf16_t)W[n * K + k] : (bf16_t)0.f;
  }
}

__global__ void zero_f32(float4* __restrict__ p, int n4) {
  const float4 z = make_float4(0.f, 0.f, 0.f, 0.f);
  for (int i = blockIdx.x * blockDim.x + threadIdx.x; i < n4;
       i += gridDim.x * blockDim.x)
    p[i] = z;
}

// ---------------------------------------------------------------------------
// Edge stage: one block = 16 edges, 8 waves; wave w owns hidden cols [16w,16w+16).
// ---------------------------------------------------------------------------
__global__ __launch_bounds__(256)
void edge_mlp_scatter(const float* __restrict__ x,
                      const int*   __restrict__ edge_index,  // [2,E]
                      const float* __restrict__ edge_attr,   // [E,ED]
                      const float* __restrict__ b1,
                      const float* __restrict__ b2,
                      const bf16_t* __restrict__ W1p,        // 8 nt x 5 kc frags
                      const bf16_t* __restrict__ W2p,        // 8 nt x 4 kc frags
                      float* __restrict__ agg,               // [N,ND]
                      int E) {
  __shared__ __align__(16) bf16_t Ain[16][AS_IN];
  __shared__ __align__(16) bf16_t H1[16][AS_H];
  __shared__ int sSrc[16];
  __shared__ int sDst[16];

  const int tid  = threadIdx.x;
  const int lane = tid & 31;
  const int w    = tid >> 5;
  const int e0   = blockIdx.x * 16;

  if (tid < 16) {
    const int e = e0 + tid;
    sSrc[tid] = (e < E) ? edge_index[e] : -1;
    sDst[tid] = (e < E) ? edge_index[E + e] : -1;
  }
  __syncthreads();

  // Gather [x[src] || edge_attr] -> bf16 LDS A tile (16 x 160), float4 loads.
  const float4* __restrict__ x4  = (const float4*)x;          // [N][32]
  const float4* __restrict__ ea4 = (const float4*)edge_attr;  // [E][8]
  for (int idx = tid; idx < 16 * (KIN / 4); idx += 256) {
    const int r  = idx / (KIN / 4);
    const int k4 = idx - r * (KIN / 4);
    const int s  = sSrc[r];
    float4 v = make_float4(0.f, 0.f, 0.f, 0.f);
    if (s >= 0)
      v = (k4 < ND / 4) ? x4[s * (ND / 4) + k4]
                        : ea4[(e0 + r) * (ED / 4) + (k4 - ND / 4)];
    *(v4bf*)&Ain[r][k4 * 4] = cvt4(v);
  }
  __syncthreads();

  // Layer 1: h1 = silu(A @ W1^T + b1), K = 160 = 5 chunks.
  v8f acc = {};
#pragma unroll
  for (int kc = 0; kc < 5; ++kc) {
    v16bf a = load_afrag(&Ain[0][0], AS_IN, lane, kc);
    v16bf b = *(const v16bf*)(W1p + ((w * 5 + kc) * 32 + lane) * 16);
    acc = wmma_bf16(a, b, acc);
  }
  {
    const int n     = w * 16 + (lane & 15);
    const int mbase = (lane >> 4) * 8;       // C layout: lanes 16-31 hold M+8
    const float bias = b1[n];
#pragma unroll
    for (int r = 0; r < 8; ++r)
      H1[mbase + r][n] = (bf16_t)fast_silu(acc[r] + bias);
  }
  __syncthreads();

  // Layer 2: m = silu(h1 @ W2^T + b2), K = 128 = 4 chunks, then scatter-add.
  v8f acc2 = {};
#pragma unroll
  for (int kc = 0; kc < 4; ++kc) {
    v16bf a = load_afrag(&H1[0][0], AS_H, lane, kc);
    v16bf b = *(const v16bf*)(W2p + ((w * 4 + kc) * 32 + lane) * 16);
    acc2 = wmma_bf16(a, b, acc2);
  }
  {
    const int n     = w * 16 + (lane & 15);
    const int mbase = (lane >> 4) * 8;
    const float bias = b2[n];
#pragma unroll
    for (int r = 0; r < 8; ++r) {
      const float v = fast_silu(acc2[r] + bias);
      const int d = sDst[mbase + r];
      if (d >= 0) unsafeAtomicAdd(&agg[d * ND + n], v);  // global_atomic_add_f32
    }
  }
}

// ---------------------------------------------------------------------------
// GRU stage: one block = 16 nodes, 8 waves; wave w owns cols [16w,16w+16) of
// each of the three gates (r,z,n) -> 6 accumulators, 24 WMMAs, fused epilogue.
// ---------------------------------------------------------------------------
__global__ __launch_bounds__(256)
void gru_update(const float* __restrict__ x,
                const float* __restrict__ b_ih,
                const float* __restrict__ b_hh,
                const bf16_t* __restrict__ Wihp,   // 24 nt x 4 kc frags
                const bf16_t* __restrict__ Whhp,   // 24 nt x 4 kc frags
                const float* __restrict__ agg,
                float* __restrict__ out,
                int N) {
  __shared__ __align__(16) bf16_t Aa[16][AS_H];
  __shared__ __align__(16) bf16_t Xa[16][AS_H];
  __shared__ __align__(16) float  Xf[16][ND];

  const int tid  = threadIdx.x;
  const int lane = tid & 31;
  const int w    = tid >> 5;
  const int n0   = blockIdx.x * 16;

  const float4* __restrict__ x4 = (const float4*)x;    // [N][32]
  const float4* __restrict__ a4 = (const float4*)agg;  // [N][32]
  for (int idx = tid; idx < 16 * (ND / 4); idx += 256) {
    const int r  = idx >> 5;
    const int k4 = idx & 31;
    const int node = n0 + r;
    float4 av = make_float4(0.f, 0.f, 0.f, 0.f);
    float4 xv = make_float4(0.f, 0.f, 0.f, 0.f);
    if (node < N) { av = a4[node * 32 + k4]; xv = x4[node * 32 + k4]; }
    *(v4bf*)&Aa[r][k4 * 4] = cvt4(av);
    *(v4bf*)&Xa[r][k4 * 4] = cvt4(xv);
    *(float4*)&Xf[r][k4 * 4] = xv;
  }
  __syncthreads();

  v8f ir = {}, iz = {}, in_ = {}, hr = {}, hz = {}, hn = {};
#pragma unroll
  for (int kc = 0; kc < 4; ++kc) {
    v16bf aA = load_afrag(&Aa[0][0], AS_H, lane, kc);
    v16bf aX = load_afrag(&Xa[0][0], AS_H, lane, kc);
    // gate g occupies n-tiles [g*8 .. g*8+7]; this wave takes nt = g*8 + w.
    v16bf b;
    b = *(const v16bf*)(Wihp + (((0 * 8 + w) * 4 + kc) * 32 + lane) * 16);
    ir = wmma_bf16(aA, b, ir);
    b = *(const v16bf*)(Wihp + (((1 * 8 + w) * 4 + kc) * 32 + lane) * 16);
    iz = wmma_bf16(aA, b, iz);
    b = *(const v16bf*)(Wihp + (((2 * 8 + w) * 4 + kc) * 32 + lane) * 16);
    in_ = wmma_bf16(aA, b, in_);
    b = *(const v16bf*)(Whhp + (((0 * 8 + w) * 4 + kc) * 32 + lane) * 16);
    hr = wmma_bf16(aX, b, hr);
    b = *(const v16bf*)(Whhp + (((1 * 8 + w) * 4 + kc) * 32 + lane) * 16);
    hz = wmma_bf16(aX, b, hz);
    b = *(const v16bf*)(Whhp + (((2 * 8 + w) * 4 + kc) * 32 + lane) * 16);
    hn = wmma_bf16(aX, b, hn);
  }

  const int n     = w * 16 + (lane & 15);
  const int mbase = (lane >> 4) * 8;
  const float bir = b_ih[n],          bhr = b_hh[n];
  const float biz = b_ih[ND + n],     bhz = b_hh[ND + n];
  const float bin = b_ih[2 * ND + n], bhn = b_hh[2 * ND + n];
#pragma unroll
  for (int r = 0; r < 8; ++r) {
    const int m = mbase + r;
    const int node = n0 + m;
    if (node >= N) continue;
    const float rg = fast_sigmoid((ir[r] + bir) + (hr[r] + bhr));
    const float zg = fast_sigmoid((iz[r] + biz) + (hz[r] + bhz));
    const float ng = fast_tanh((in_[r] + bin) + rg * (hn[r] + bhn));
    const float xv = Xf[m][n];
    out[node * ND + n] = (1.f - zg) * ng + zg * xv;
  }
}

// ---------------------------------------------------------------------------
extern "C" void kernel_launch(void* const* d_in, const int* in_sizes, int n_in,
                              void* d_out, int out_size, void* d_ws, size_t ws_size,
                              hipStream_t stream) {
  const float* x   = (const float*)d_in[0];
  const int*   ei  = (const int*)d_in[1];
  const float* ea  = (const float*)d_in[2];
  const float* W1  = (const float*)d_in[3];
  const float* b1  = (const float*)d_in[4];
  const float* W2  = (const float*)d_in[5];
  const float* b2  = (const float*)d_in[6];
  const float* Wih = (const float*)d_in[7];
  const float* bih = (const float*)d_in[8];
  const float* Whh = (const float*)d_in[9];
  const float* bhh = (const float*)d_in[10];
  float* out = (float*)d_out;

  const int N = in_sizes[0] / ND;
  const int E = in_sizes[1] / 2;

  // Workspace layout (bytes): W1p 40960 | W2p 32768 | Wihp 98304 | Whhp 98304
  // | agg N*128*4 (~25.6 MB). Total < 26 MB.
  const int nW1  = 8 * 5 * 512;    // 20480 frag elems
  const int nW2  = 8 * 4 * 512;    // 16384
  const int nWg  = 24 * 4 * 512;   // 49152 (each of W_ih, W_hh)
  bf16_t* W1p  = (bf16_t*)d_ws;
  bf16_t* W2p  = W1p + nW1;
  bf16_t* Wihp = W2p + nW2;
  bf16_t* Whhp = Wihp + nWg;
  float*  agg  = (float*)(Whhp + nWg);

  pack_weight_frags<<<(nW1 + 255) / 256, 256, 0, stream>>>(W1, KIN, 5, nW1, W1p);
  pack_weight_frags<<<(nW2 + 255) / 256, 256, 0, stream>>>(W2, HID, 4, nW2, W2p);
  pack_weight_frags<<<(nWg + 255) / 256, 256, 0, stream>>>(Wih, HID, 4, nWg, Wihp);
  pack_weight_frags<<<(nWg + 255) / 256, 256, 0, stream>>>(Whh, ND, 4, nWg, Whhp);

  zero_f32<<<1024, 256, 0, stream>>>((float4*)agg, N * ND / 4);

  edge_mlp_scatter<<<(E + 15) / 16, 256, 0, stream>>>(x, ei, ea, b1, b2, W1p, W2p,
                                                      agg, E);
  gru_update<<<(N + 15) / 16, 256, 0, stream>>>(x, bih, bhh, Wihp, Whhp, agg, out, N);
}
